// Score_Func_7258494730904
// MI455X (gfx1250) — matrix operand order repaired and checked by reading, not compiled, not run on hardware
//
#include <hip/hip_runtime.h>
#include <hip/hip_bf16.h>
#include <math.h>

#define BATCHES 64
#define NPTS    1024
#define NDIM    32
#define EPS_T   0.0025f                  // blur^p = 0.05^2
#define INV_EPS 400.0f
#define N_ITER  40
#define L2E     1.4426950408889634f      // log2(e)
#define LN2     0.6931471805599453f
#define QSCALE  (INV_EPS * L2E)          // dot -> log2-domain score
#define LOGW2   (-10.0f)                 // log2(1/1024)
#define EPSLN2  (EPS_T * LN2)            // converts log2-lse back: -eps*ln(sum)

typedef __attribute__((ext_vector_type(16))) _Float16 v16h;
typedef __attribute__((ext_vector_type(8)))  float    v8f;

union FragU { v16h h; uint4 u[2]; };

__device__ __forceinline__ float exp2_fast(float x) { return __builtin_amdgcn_exp2f(x); }
__device__ __forceinline__ float log2_fast(float x) { return __builtin_amdgcn_logf(x);  }

// Flash-softmax accumulation of one 16x16 WMMA tile (8 rows of one column per lane),
// entirely in the log2 domain. alpha values are pre-scaled by log2(e).
__device__ __forceinline__ void online_update(const v8f& acc,
                                              const float4& a0, const float4& a1,
                                              float& m, float& s) {
  float sc[8];
  sc[0] = fmaf(acc[0], QSCALE, a0.x);
  sc[1] = fmaf(acc[1], QSCALE, a0.y);
  sc[2] = fmaf(acc[2], QSCALE, a0.z);
  sc[3] = fmaf(acc[3], QSCALE, a0.w);
  sc[4] = fmaf(acc[4], QSCALE, a1.x);
  sc[5] = fmaf(acc[5], QSCALE, a1.y);
  sc[6] = fmaf(acc[6], QSCALE, a1.z);
  sc[7] = fmaf(acc[7], QSCALE, a1.w);
  float tm = fmaxf(fmaxf(fmaxf(sc[0], sc[1]), fmaxf(sc[2], sc[3])),
                   fmaxf(fmaxf(sc[4], sc[5]), fmaxf(sc[6], sc[7])));
  const float nm = fmaxf(m, tm);
  float sum = 0.0f;
  #pragma unroll
  for (int r = 0; r < 8; ++r) sum += exp2_fast(sc[r] - nm);
  s = fmaf(s, exp2_fast(m - nm), sum);   // exp2(-inf)=0 handles the first tile
  m = nm;
}

// Combine the two half-wave partials (lanes L and L+16 hold the same column),
// return lse in log2 domain: max + log2(sum).
__device__ __forceinline__ float finish_col(float m, float s) {
  const float mo = __shfl_xor(m, 16, 32);
  const float so = __shfl_xor(s, 16, 32);
  const float nm = fmaxf(m, mo);
  const float st = fmaf(so, exp2_fast(mo - nm), s * exp2_fast(m - nm));
  return nm + log2_fast(st);
}

// Generic softmin pass:
//   out_j = -eps*ln2 * lse2_over_rows( alphaIn_row + dot(row,col)*400*log2e ) + colNorm_j
//   (avgMode: out_j = 0.5*(prev_j + out_j));  alphaOut_j = (out_j - colNorm_j)*QSCALE - 10
// Each wave owns TWO 16-column tiles (32 cols) and streams 64 row tiles.
__global__ __launch_bounds__(128) void softmin_pass(
    float* __restrict__ out, float* __restrict__ alphaOut,
    const float* __restrict__ prev,
    const _Float16* __restrict__ rowH, const _Float16* __restrict__ rowL,
    const float* __restrict__ alphaIn,
    const _Float16* __restrict__ colH, const _Float16* __restrict__ colL,
    const float* __restrict__ colNorm, int avgMode)
{
  const int  b        = blockIdx.y;
  const int  lane     = threadIdx.x & 31;
  const int  wave     = threadIdx.x >> 5;
  const int  colBase0 = (blockIdx.x * 4 + wave) * 32;   // 32 columns per wave
  const int  colBase1 = colBase0 + 16;
  const size_t pbase  = (size_t)b * NPTS * NDIM;
  const int  vbase    = b * NPTS;
  const bool hi       = lane >= 16;
  const int  l15      = lane & 15;

  // Uniform base pointers (SGPRs) + 32-bit byte offsets (VGPR) -> saddr loads.
  const char* colHB = (const char*)(colH + pbase);
  const char* colLB = (const char*)(colL + pbase);
  const char* rowHB = (const char*)(rowH + pbase);
  const char* rowLB = (const char*)(rowL + pbase);
  const char* aInB  = (const char*)(alphaIn + vbase);

  // B fragment (32x16 f16): col = base + l15, 16 contiguous K per half-wave.
  const int bOff0 = (colBase0 + l15) * (NDIM * 2) + (hi ? 32 : 0);
  const int bOff1 = bOff0 + 16 * (NDIM * 2);
  v16h bh0, bl0, bh1, bl1;
  { FragU f; f.u[0] = *(const uint4*)(colHB + bOff0); f.u[1] = *(const uint4*)(colHB + bOff0 + 16); bh0 = f.h; }
  { FragU f; f.u[0] = *(const uint4*)(colLB + bOff0); f.u[1] = *(const uint4*)(colLB + bOff0 + 16); bl0 = f.h; }
  { FragU f; f.u[0] = *(const uint4*)(colHB + bOff1); f.u[1] = *(const uint4*)(colHB + bOff1 + 16); bh1 = f.h; }
  { FragU f; f.u[0] = *(const uint4*)(colLB + bOff1); f.u[1] = *(const uint4*)(colLB + bOff1 + 16); bl1 = f.h; }

  // A fragment lane offset (16x32 f16): row = base + l15, K={0..15}/{16..31}.
  const int laneA = l15 * (NDIM * 2) + (hi ? 32 : 0);

  float m0 = -INFINITY, s0 = 0.0f;
  float m1 = -INFINITY, s1 = 0.0f;

  int rowOff = 0;              // byte offset into point arrays (16 rows = 1024 B)
  int aOff   = hi ? 32 : 0;    // byte offset into alpha array  (16 rows = 64 B)
  for (int rt = 0; rt < NPTS / 16; ++rt, rowOff += 16 * NDIM * 2, aOff += 16 * 4) {
    if (rt + 1 < NPTS / 16)    // prefetch next A tile (global_prefetch_b8)
      __builtin_prefetch(rowHB + rowOff + 16 * NDIM * 2 + l15 * (NDIM * 2), 0, 1);

    FragU fah, fal;
    fah.u[0] = *(const uint4*)(rowHB + rowOff + laneA);
    fah.u[1] = *(const uint4*)(rowHB + rowOff + laneA + 16);
    fal.u[0] = *(const uint4*)(rowLB + rowOff + laneA);
    fal.u[1] = *(const uint4*)(rowLB + rowOff + laneA + 16);

    // fp32-accurate Gram tiles via f16 hi/lo split; interleave the two
    // accumulators so dependent WMMAs never issue back-to-back.
    v8f acc0 = {}, acc1 = {};
    acc0 = __builtin_amdgcn_wmma_f32_16x16x32_f16(false, fal.h, false, bh0, (short)0, acc0, false, false);
    acc1 = __builtin_amdgcn_wmma_f32_16x16x32_f16(false, fal.h, false, bh1, (short)0, acc1, false, false);
    acc0 = __builtin_amdgcn_wmma_f32_16x16x32_f16(false, fah.h, false, bl0, (short)0, acc0, false, false);
    acc1 = __builtin_amdgcn_wmma_f32_16x16x32_f16(false, fah.h, false, bl1, (short)0, acc1, false, false);
    acc0 = __builtin_amdgcn_wmma_f32_16x16x32_f16(false, fah.h, false, bh0, (short)0, acc0, false, false);
    acc1 = __builtin_amdgcn_wmma_f32_16x16x32_f16(false, fah.h, false, bh1, (short)0, acc1, false, false);

    // alpha for rows rOff..rOff+7 (log2-scaled), shared by both column tiles
    const float4 a0 = *(const float4*)(aInB + aOff);
    const float4 a1 = *(const float4*)(aInB + aOff + 16);

    online_update(acc0, a0, a1, m0, s0);
    online_update(acc1, a0, a1, m1, s1);
  }

  const float lse0 = finish_col(m0, s0);
  const float lse1 = finish_col(m1, s1);

  if (!hi) {
    const int j0 = vbase + colBase0 + lane;
    const int j1 = vbase + colBase1 + lane;
    const float cn0 = colNorm[j0];
    const float cn1 = colNorm[j1];
    float v0 = fmaf(-EPSLN2, lse0, cn0);
    float v1 = fmaf(-EPSLN2, lse1, cn1);
    if (avgMode) {
      v0 = 0.5f * (prev[j0] + v0);
      v1 = 0.5f * (prev[j1] + v1);
    }
    out[j0] = v0;
    out[j1] = v1;
    alphaOut[j0] = fmaf(v0 - cn0, QSCALE, LOGW2);
    alphaOut[j1] = fmaf(v1 - cn1, QSCALE, LOGW2);
  }
}

// Split f32 points into f16 hi/lo pair and compute half squared norms.
__global__ __launch_bounds__(256) void prep_split(
    const float* __restrict__ X, const float* __restrict__ Y,
    _Float16* __restrict__ Xh, _Float16* __restrict__ Xl,
    _Float16* __restrict__ Yh, _Float16* __restrict__ Yl,
    float* __restrict__ hx, float* __restrict__ hy)
{
  const int total = BATCHES * NPTS;
  int idx = blockIdx.x * blockDim.x + threadIdx.x;
  if (idx >= 2 * total) return;
  const bool isY = idx >= total;
  const int  p   = isY ? idx - total : idx;
  const float* src = (isY ? Y : X) + (size_t)p * NDIM;
  _Float16* dh = (isY ? Yh : Xh) + (size_t)p * NDIM;
  _Float16* dl = (isY ? Yl : Xl) + (size_t)p * NDIM;
  float nrm = 0.0f;
  #pragma unroll
  for (int k = 0; k < NDIM; ++k) {
    const float v = src[k];
    nrm = fmaf(v, v, nrm);
    const _Float16 h = (_Float16)v;
    dh[k] = h;
    dl[k] = (_Float16)(v - (float)h);
  }
  (isY ? hy : hx)[p] = 0.5f * nrm;
}

// Initial log2-domain alphas for f=g=0, and zero the symmetric potentials.
__global__ __launch_bounds__(256) void init_state(
    const float* __restrict__ hx, const float* __restrict__ hy,
    float* __restrict__ aXxy, float* __restrict__ aXxx0,
    float* __restrict__ aYyy0, float* __restrict__ fxx0, float* __restrict__ gyy0)
{
  const int i = blockIdx.x * blockDim.x + threadIdx.x;
  if (i >= BATCHES * NPTS) return;
  const float ax = fmaf(-hx[i], QSCALE, LOGW2);
  const float ay = fmaf(-hy[i], QSCALE, LOGW2);
  aXxy[i]  = ax;
  aXxx0[i] = ax;
  aYyy0[i] = ay;
  fxx0[i]  = 0.0f;
  gyy0[i]  = 0.0f;
}

// S_b = mean(f_xy - f_xx) + mean(g_xy - g_yy)
__global__ __launch_bounds__(256) void finalize(
    float* __restrict__ out,
    const float* __restrict__ fxy, const float* __restrict__ fxx,
    const float* __restrict__ gxy, const float* __restrict__ gyy)
{
  __shared__ float sm[256];
  const int b = blockIdx.x;
  const int base = b * NPTS;
  float acc = 0.0f;
  for (int i = threadIdx.x; i < NPTS; i += 256)
    acc += (fxy[base + i] - fxx[base + i]) + (gxy[base + i] - gyy[base + i]);
  sm[threadIdx.x] = acc;
  __syncthreads();
  for (int off = 128; off > 0; off >>= 1) {
    if (threadIdx.x < off) sm[threadIdx.x] += sm[threadIdx.x + off];
    __syncthreads();
  }
  if (threadIdx.x == 0) out[b] = sm[0] * (1.0f / NPTS);
}

extern "C" void kernel_launch(void* const* d_in, const int* in_sizes, int n_in,
                              void* d_out, int out_size, void* d_ws, size_t ws_size,
                              hipStream_t stream) {
  const float* X = (const float*)d_in[0];
  const float* Y = (const float*)d_in[1];
  float* out = (float*)d_out;

  char* ws = (char*)d_ws;
  size_t off = 0;
  auto alloc = [&](size_t bytes) -> void* {
    void* p = ws + off;
    off += (bytes + 255) & ~(size_t)255;
    return p;
  };
  const size_t PTS_BYTES = (size_t)BATCHES * NPTS * NDIM * sizeof(_Float16); // 4 MB
  const size_t VEC_BYTES = (size_t)BATCHES * NPTS * sizeof(float);           // 256 KB

  _Float16* Xh = (_Float16*)alloc(PTS_BYTES);
  _Float16* Xl = (_Float16*)alloc(PTS_BYTES);
  _Float16* Yh = (_Float16*)alloc(PTS_BYTES);
  _Float16* Yl = (_Float16*)alloc(PTS_BYTES);
  float* hx   = (float*)alloc(VEC_BYTES);
  float* hy   = (float*)alloc(VEC_BYTES);
  float* fxy  = (float*)alloc(VEC_BYTES);
  float* gxy  = (float*)alloc(VEC_BYTES);
  float* aXxy = (float*)alloc(VEC_BYTES);
  float* aYxy = (float*)alloc(VEC_BYTES);
  float* fxx[2]  = { (float*)alloc(VEC_BYTES), (float*)alloc(VEC_BYTES) };
  float* aXxx[2] = { (float*)alloc(VEC_BYTES), (float*)alloc(VEC_BYTES) };
  float* gyy[2]  = { (float*)alloc(VEC_BYTES), (float*)alloc(VEC_BYTES) };
  float* aYyy[2] = { (float*)alloc(VEC_BYTES), (float*)alloc(VEC_BYTES) };

  const int total = BATCHES * NPTS;
  prep_split<<<(2 * total + 255) / 256, 256, 0, stream>>>(X, Y, Xh, Xl, Yh, Yl, hx, hy);
  init_state<<<(total + 255) / 256, 256, 0, stream>>>(hx, hy, aXxy, aXxx[0], aYyy[0], fxx[0], gyy[0]);

  const dim3 grid(NPTS / 128, BATCHES);   // 4 waves per WG, 32 cols per wave
  int p = 0;
  for (int t = 0; t < N_ITER; ++t) {
    // g_xy from f_xy (rows = X, cols = Y)
    softmin_pass<<<grid, 128, 0, stream>>>(gxy, aYxy, nullptr,
                                           Xh, Xl, aXxy, Yh, Yl, hy, 0);
    // symmetric fixed point on X (averaged, ping-pong)
    softmin_pass<<<grid, 128, 0, stream>>>(fxx[1 - p], aXxx[1 - p], fxx[p],
                                           Xh, Xl, aXxx[p], Xh, Xl, hx, 1);
    // symmetric fixed point on Y
    softmin_pass<<<grid, 128, 0, stream>>>(gyy[1 - p], aYyy[1 - p], gyy[p],
                                           Yh, Yl, aYyy[p], Yh, Yl, hy, 1);
    // f_xy from the freshly updated g_xy (rows = Y, cols = X)
    softmin_pass<<<grid, 128, 0, stream>>>(fxy, aXxy, nullptr,
                                           Yh, Yl, aYxy, Xh, Xl, hx, 0);
    p ^= 1;
  }
  // after an even number of iterations the symmetric results are in buffer 0
  finalize<<<BATCHES, 256, 0, stream>>>(out, fxy, fxx[0], gxy, gyy[0]);
}
